// GraphGenLSTMGenerator_14070312861696
// MI455X (gfx1250) — compile-verified, matrix operand order checked by reading, not run on hardware
//
#include <hip/hip_runtime.h>

#define NN     64
#define N2     4096
#define STEPS  63
#define BSZ    256
#define H4     256        // 4*N gate width

typedef __bf16 bf16_t;
typedef __bf16 v16bf  __attribute__((ext_vector_type(16)));
typedef float  v8f    __attribute__((ext_vector_type(8)));
typedef unsigned int u32x4 __attribute__((ext_vector_type(4)));

union FragU { u32x4 q[2]; v16bf v; };

// Load a 16x32 bf16 WMMA A/B fragment from a row-major bf16 matrix.
// Per ISA layout (16-bit A 16x32): lane<16 holds row (lane&15), k = e0..e0+7 (v0..3)
// and e0+16..e0+23 (v4..7); lanes>=16 add +8 to k. Both halves are contiguous 16B.
__device__ __forceinline__ v16bf load_frag(const bf16_t* __restrict__ base,
                                           int row, int ld, int e0) {
  FragU f;
  const bf16_t* p = base + (size_t)row * (size_t)ld + e0;
  f.q[0] = *(const u32x4*)(p);
  f.q[1] = *(const u32x4*)(p + 16);
  return f.v;
}

// Same fragment, but from an LDS tile stored in fragment-major layout:
// tile j occupies 512 bf16 = [lane][16 elems]; per-lane chunk is 32 contiguous B.
__device__ __forceinline__ v16bf lds_frag(const bf16_t* p) {
  FragU f;
  f.q[0] = *(const u32x4*)(p);
  f.q[1] = *(const u32x4*)(p + 8);
  return f.v;
}

__device__ __forceinline__ v8f wmma_bf16(v16bf a, v16bf b, v8f c) {
  return __builtin_amdgcn_wmma_f32_16x16x32_bf16(false, a, false, b, (short)0, c,
                                                 false, false);
}

__device__ __forceinline__ float sigmoidf_(float x) {
  return 1.0f / (1.0f + __expf(-x));
}

// ---------------- weight conversion (fp32 -> bf16 row-major) ----------------
__global__ void cvt_wih(const float* __restrict__ src, bf16_t* __restrict__ dst) {
  int i = blockIdx.x * blockDim.x + threadIdx.x;          // H4 * N2 threads
  if (i >= H4 * N2) return;
  int r = i >> 12, k = i & (N2 - 1);
  dst[i] = (bf16_t)src[(size_t)r * (N2 + 1) + k];         // drop noise column
}

__global__ void cvt_plain(const float* __restrict__ src, bf16_t* __restrict__ dst,
                          int n) {
  int i = blockIdx.x * blockDim.x + threadIdx.x;
  if (i < n) dst[i] = (bf16_t)src[i];
}

// ---------------- step kernel 1: gates = pred@Wih^T + h@Whh^T + bias --------
__global__ void gates_wmma(const bf16_t* __restrict__ pred_bf,
                           const bf16_t* __restrict__ Wih_bf,
                           const bf16_t* __restrict__ h_in,
                           const bf16_t* __restrict__ Whh_bf,
                           const float*  __restrict__ Wih_f32,
                           const float*  __restrict__ bih,
                           const float*  __restrict__ bhh,
                           const float*  __restrict__ noise,
                           float* __restrict__ gates) {
  int wave = (blockIdx.x * blockDim.x + threadIdx.x) >> 5;
  int lane = threadIdx.x & 31;
  int mT = wave >> 4, nT = wave & 15;
  int lr = lane & 15;
  int kh = (lane >> 4) << 3;
  int arow = mT * 16 + lr;
  int brow = nT * 16 + lr;

  v8f acc = {};
#pragma unroll 4
  for (int kT = 0; kT < N2 / 32; ++kT) {
    int e0 = kT * 32 + kh;
    v16bf a = load_frag(pred_bf, arow, N2, e0);
    v16bf b = load_frag(Wih_bf, brow, N2, e0);
    acc = wmma_bf16(a, b, acc);
  }
#pragma unroll
  for (int kT = 0; kT < NN / 32; ++kT) {
    int e0 = kT * 32 + kh;
    v16bf a = load_frag(h_in, arow, NN, e0);
    v16bf b = load_frag(Whh_bf, brow, NN, e0);
    acc = wmma_bf16(a, b, acc);
  }

  int rc = brow;                                   // gate column (0..255)
  float bias = bih[rc] + bhh[rc];
  float wn = Wih_f32[(size_t)rc * (N2 + 1) + N2];  // noise-column weight
  int mhi = (lane >> 4) << 3;
#pragma unroll
  for (int r = 0; r < 8; ++r) {
    int b = mT * 16 + mhi + r;
    gates[(size_t)b * H4 + rc] = acc[r] + bias + noise[b] * wn;
  }
}

// ---------------- step kernel 2: LSTM cell ----------------------------------
__global__ void lstm_cell(const float* __restrict__ gates,
                          const float* __restrict__ c_in,
                          float* __restrict__ c_out,
                          bf16_t* __restrict__ h_out_bf) {
  int idx = blockIdx.x * blockDim.x + threadIdx.x;  // BSZ*NN threads
  int b = idx >> 6, j = idx & 63;
  const float* g = gates + (size_t)b * H4;
  float ig = sigmoidf_(g[j]);
  float fg = sigmoidf_(g[64 + j]);
  float gg = tanhf(g[128 + j]);
  float og = sigmoidf_(g[192 + j]);
  float c  = fg * c_in[idx] + ig * gg;
  float h  = og * tanhf(c);
  c_out[idx]   = c;
  h_out_bf[idx] = (bf16_t)h;
}

// ---------------- step kernel 3: mid = tanh(h@W1^T + b1) --------------------
// 512 waves; wave -> (mG covering 2 M-tiles, nG covering 4 N-tiles)
__global__ void mid_wmma(const bf16_t* __restrict__ h_bf,
                         const bf16_t* __restrict__ W1_bf,
                         const float*  __restrict__ b1,
                         bf16_t* __restrict__ mid_bf) {
  int wave = (blockIdx.x * blockDim.x + threadIdx.x) >> 5;
  int lane = threadIdx.x & 31;
  int mG = wave >> 6, nG = wave & 63;
  int lr = lane & 15;
  int kh = (lane >> 4) << 3;
  int arow0 = (mG * 2) * 16 + lr;

  v8f acc[2][4] = {};
#pragma unroll
  for (int kT = 0; kT < NN / 32; ++kT) {
    int e0 = kT * 32 + kh;
    v16bf a0 = load_frag(h_bf, arow0,      NN, e0);
    v16bf a1 = load_frag(h_bf, arow0 + 16, NN, e0);
#pragma unroll
    for (int j = 0; j < 4; ++j) {
      v16bf b = load_frag(W1_bf, (nG * 4 + j) * 16 + lr, NN, e0);
      acc[0][j] = wmma_bf16(a0, b, acc[0][j]);
      acc[1][j] = wmma_bf16(a1, b, acc[1][j]);
    }
  }
  int mhi = (lane >> 4) << 3;
#pragma unroll
  for (int mi = 0; mi < 2; ++mi) {
#pragma unroll
    for (int j = 0; j < 4; ++j) {
      int col = (nG * 4 + j) * 16 + lr;
      float bb = b1[col];
#pragma unroll
      for (int r = 0; r < 8; ++r) {
        int b = (mG * 2 + mi) * 16 + mhi + r;
        mid_bf[(size_t)b * N2 + col] = (bf16_t)tanhf(acc[mi][j][r] + bb);
      }
    }
  }
}

// ---------------- step kernel 4: p = sigmoid(mid@W2^T + b2) -----------------
// Dominant GEMM: 256x4096x4096. Block = 8 waves = 8 M-groups (full M=256) x one
// 64-column N-group. All waves share the same 4 W2 B-tiles per k-step, staged
// cooperatively through double-buffered LDS -> W2 is read from L2 exactly once
// per step (32 MB) instead of 8x.
__global__ void out_wmma(const bf16_t* __restrict__ mid_bf,
                         const bf16_t* __restrict__ W2_bf,
                         const float*  __restrict__ b2,
                         float* __restrict__ out,
                         bf16_t* __restrict__ pred_bf,
                         int t) {
  __shared__ __align__(16) bf16_t Bsh[2][4 * 32 * 16];   // 2 x 4KB

  int nG   = blockIdx.x;              // 0..63  (column group of 4 N-tiles)
  int mG   = threadIdx.x >> 5;        // 0..7   (wave = M-group of 2 M-tiles)
  int lane = threadIdx.x & 31;
  int lr   = lane & 15;
  int kh   = (lane >> 4) << 3;
  int arow0 = (mG * 2) * 16 + lr;

  // Cooperative B-chunk mapping: thread -> (tile j, lane l, half hf)
  int cj  = threadIdx.x >> 6;         // 0..3
  int cl  = (threadIdx.x >> 1) & 31;  // 0..31
  int chf = threadIdx.x & 1;          // 0..1
  const bf16_t* brow_p =
      W2_bf + (size_t)(nG * 64 + cj * 16 + (cl & 15)) * N2 + ((cl >> 4) << 3) +
      (chf << 4);
  bf16_t* bsto0 = &Bsh[0][cj * 512 + cl * 16 + chf * 8];
  bf16_t* bsto1 = &Bsh[1][cj * 512 + cl * 16 + chf * 8];

  v8f acc[2][4] = {};
  u32x4 breg = *(const u32x4*)(brow_p);               // k-tile 0 chunk

  for (int kT = 0; kT < N2 / 32; ++kT) {
    if (kT) __syncthreads();                          // prior reads of buf done
    bf16_t* sto = (kT & 1) ? bsto1 : bsto0;
    *(u32x4*)sto = breg;
    if (kT + 1 < N2 / 32)
      breg = *(const u32x4*)(brow_p + (kT + 1) * 32); // prefetch next chunk
    __syncthreads();                                  // staged tile visible

    const bf16_t* buf = (kT & 1) ? Bsh[1] : Bsh[0];
    int e0 = kT * 32 + kh;
    v16bf a0 = load_frag(mid_bf, arow0,      N2, e0);
    v16bf a1 = load_frag(mid_bf, arow0 + 16, N2, e0);
#pragma unroll
    for (int j = 0; j < 4; ++j) {
      v16bf b = lds_frag(buf + j * 512 + lane * 16);
      acc[0][j] = wmma_bf16(a0, b, acc[0][j]);
      acc[1][j] = wmma_bf16(a1, b, acc[1][j]);
    }
  }

  int mhi = (lane >> 4) << 3;
#pragma unroll
  for (int mi = 0; mi < 2; ++mi) {
#pragma unroll
    for (int j = 0; j < 4; ++j) {
      int col = (nG * 4 + j) * 16 + lr;
      float bb = b2[col];
#pragma unroll
      for (int r = 0; r < 8; ++r) {
        int b = (mG * 2 + mi) * 16 + mhi + r;
        float p = sigmoidf_(acc[mi][j][r] + bb);
        out[(size_t)b * (STEPS * N2) + (size_t)t * N2 + col] = p;
        pred_bf[(size_t)b * N2 + col] = (bf16_t)p;
      }
    }
  }
}

// ---------------------------------------------------------------------------
extern "C" void kernel_launch(void* const* d_in, const int* in_sizes, int n_in,
                              void* d_out, int out_size, void* d_ws, size_t ws_size,
                              hipStream_t stream) {
  const float* noise = (const float*)d_in[0];   // [256,1]
  const float* Wih   = (const float*)d_in[1];   // [256,4097]
  const float* Whh   = (const float*)d_in[2];   // [256,64]
  const float* bih   = (const float*)d_in[3];   // [256]
  const float* bhh   = (const float*)d_in[4];   // [256]
  const float* W1    = (const float*)d_in[5];   // [4096,64]
  const float* b1    = (const float*)d_in[6];   // [4096]
  const float* W2    = (const float*)d_in[7];   // [4096,4096]
  const float* b2    = (const float*)d_in[8];   // [4096]
  float* out = (float*)d_out;

  char* ws = (char*)d_ws;
  size_t off = 0;
  bf16_t* W2b   = (bf16_t*)(ws + off); off += (size_t)N2 * N2 * 2;      // 32 MB
  bf16_t* Wihb  = (bf16_t*)(ws + off); off += (size_t)H4 * N2 * 2;      //  2 MB
  bf16_t* W1b   = (bf16_t*)(ws + off); off += (size_t)N2 * NN * 2;      // 512 KB
  bf16_t* Whhb  = (bf16_t*)(ws + off); off += (size_t)H4 * NN * 2;      //  32 KB
  bf16_t* predb = (bf16_t*)(ws + off); off += (size_t)BSZ * N2 * 2;     //  2 MB
  bf16_t* midb  = (bf16_t*)(ws + off); off += (size_t)BSZ * N2 * 2;     //  2 MB
  bf16_t* hbuf0 = (bf16_t*)(ws + off); off += (size_t)BSZ * NN * 2;
  bf16_t* hbuf1 = (bf16_t*)(ws + off); off += (size_t)BSZ * NN * 2;
  float*  cbuf0 = (float*) (ws + off); off += (size_t)BSZ * NN * 4;
  float*  cbuf1 = (float*) (ws + off); off += (size_t)BSZ * NN * 4;
  float*  gates = (float*) (ws + off); off += (size_t)BSZ * H4 * 4;

  // Initial state: pred0 = h0 = c0 = 0
  hipMemsetAsync(predb, 0, (size_t)BSZ * N2 * 2, stream);
  hipMemsetAsync(hbuf0, 0, (size_t)BSZ * NN * 2, stream);
  hipMemsetAsync(cbuf0, 0, (size_t)BSZ * NN * 4, stream);

  // One-time bf16 weight conversion (re-done each call; deterministic)
  cvt_wih  <<<(H4 * N2 + 255) / 256, 256, 0, stream>>>(Wih, Wihb);
  cvt_plain<<<(H4 * NN + 255) / 256, 256, 0, stream>>>(Whh, Whhb, H4 * NN);
  cvt_plain<<<(N2 * NN + 255) / 256, 256, 0, stream>>>(W1, W1b, N2 * NN);
  cvt_plain<<<(N2 * N2 + 255) / 256, 256, 0, stream>>>(W2, W2b, N2 * N2);

  for (int t = 0; t < STEPS; ++t) {
    bf16_t* h_in  = (t & 1) ? hbuf1 : hbuf0;
    bf16_t* h_out = (t & 1) ? hbuf0 : hbuf1;
    float*  c_in  = (t & 1) ? cbuf1 : cbuf0;
    float*  c_out = (t & 1) ? cbuf0 : cbuf1;

    gates_wmma<<<64, 128, 0, stream>>>(predb, Wihb, h_in, Whhb, Wih, bih, bhh,
                                       noise, gates);
    lstm_cell<<<(BSZ * NN) / 256, 256, 0, stream>>>(gates, c_in, c_out, h_out);
    mid_wmma<<<64, 256, 0, stream>>>(h_out, W1b, b1, midb);
    out_wmma<<<64, 256, 0, stream>>>(midb, W2b, b2, out, predb, t);
  }
}
